// MultiheadAttention_90469191123215
// MI455X (gfx1250) — matrix operand order repaired and checked by reading, not compiled, not run on hardware
//
#include <hip/hip_runtime.h>
#include <hip/hip_bf16.h>

// ---------------------------------------------------------------------------
// Multi-head attention forward for MI455X (gfx1250), wave32 + WMMA.
// All GEMMs run through one templated kernel using v_wmma_f32_16x16x32_f16.
// ---------------------------------------------------------------------------

#define DM_  1024
#define H_N  16
#define DH_  64
#define B_N  2
#define S_N  2048

typedef __attribute__((ext_vector_type(16))) _Float16 v16h;
typedef __attribute__((ext_vector_type(8)))  float    v8f;

union FragAB { unsigned u[8]; v16h v; };

__device__ __forceinline__ unsigned short f2h_bits(float f) {
    union { _Float16 h; unsigned short u; } cv;
    cv.h = (_Float16)f;
    return cv.u;
}

// LDS tile strides (in ushort elements); even + padded to dodge bank conflicts.
#define AS_LD 40   // A tile: 128 rows x 32 k
#define BS_LD 40   // B tile:  64 cols x 32 k   (stored as Bs[n][k])

// MODE 0: out = acc (+ bias[col] if HAS_BIAS)
// MODE 1: out = (mask[col]==0) ? -1e-9f : acc * scale      (energy epilogue)
// BTRANS true : B source is [N,K] row-major (weights / K-projection rows)
// BTRANS false: B source is [K,N] row-major (V head) -> transposed while staging
template<int MODE, bool BTRANS, bool HAS_BIAS>
__global__ void __launch_bounds__(256)
gemm_wmma(const float* __restrict__ A,  long lda, long aSB, long aSH,
          const float* __restrict__ Bm, long ldb, long bSB, long bSH,
          float*       __restrict__ out, long ldo, long oSB, long oSH,
          const float* __restrict__ bias,
          const int*   __restrict__ mask, long mSB,
          int Kdim, float scale, int nHeads)
{
    __shared__ unsigned short As[128 * AS_LD];
    __shared__ unsigned short Bs[ 64 * BS_LD];

    const int tid  = threadIdx.x;
    const int lane = tid & 31;
    const int wave = tid >> 5;          // 8 waves
    const int waveM = wave & 3;         // 4 waves along M (32 rows each)
    const int waveN = wave >> 2;        // 2 waves along N (32 cols each)

    const int bz = blockIdx.z / nHeads;
    const int hz = blockIdx.z % nHeads;

    const float* Ap = A   + (size_t)bz * aSB + (size_t)hz * aSH;
    const float* Bp = Bm  + (size_t)bz * bSB + (size_t)hz * bSH;
    float*       Op = out + (size_t)bz * oSB + (size_t)hz * oSH;
    const int*   Mp = (MODE == 1) ? (mask + (size_t)bz * mSB) : nullptr;

    const int mBase = blockIdx.y * 128;
    const int nBase = blockIdx.x * 64;

    v8f acc[2][2];
    #pragma unroll
    for (int i = 0; i < 2; ++i)
        #pragma unroll
        for (int j = 0; j < 2; ++j)
            acc[i][j] = (v8f){0.f,0.f,0.f,0.f,0.f,0.f,0.f,0.f};

    // Per-thread staging coordinates (loop-invariant).
    const int aRow = tid >> 1;
    const int aKof = (tid & 1) * 16;
    const float* aSrc = Ap + (size_t)(mBase + aRow) * lda + aKof;

    for (int kc = 0; kc < Kdim; kc += 32) {
        // ---- stage A tile: 128 rows x 32 k, fp32 -> f16 in LDS ----
        {
            const float* src = aSrc + kc;
            // WGP-scope prefetch of next K-chunk (locality 3 -> all cache levels)
            if (kc + 32 < Kdim) __builtin_prefetch(src + 32, 0, 3);
            #pragma unroll
            for (int i = 0; i < 16; ++i)
                As[aRow * AS_LD + aKof + i] = f2h_bits(src[i]);
        }
        // ---- stage B tile into canonical Bs[n][k] ----
        if (BTRANS) {
            const int n   = tid >> 2;
            const int kof = (tid & 3) * 8;
            const float* src = Bp + (size_t)(nBase + n) * ldb + kc + kof;
            if (kc + 32 < Kdim) __builtin_prefetch(src + 32, 0, 3);
            #pragma unroll
            for (int i = 0; i < 8; ++i)
                Bs[n * BS_LD + kof + i] = f2h_bits(src[i]);
        } else {
            const int kk  = tid >> 3;
            const int nof = (tid & 7) * 8;
            const float* src = Bp + (size_t)(kc + kk) * ldb + nBase + nof;
            if (kc + 32 < Kdim) __builtin_prefetch(src + (size_t)32 * ldb, 0, 3);
            #pragma unroll
            for (int i = 0; i < 8; ++i)
                Bs[(nof + i) * BS_LD + kk] = f2h_bits(src[i]);
        }
        __syncthreads();

        // ---- load fragments (ISA 16-bit A 16x32 / B 32x16 layouts) ----
        const int mlo = lane & 15;
        const int kh  = lane >> 4;     // half-select
        FragAB afr[2], bfr[2];
        #pragma unroll
        for (int s = 0; s < 2; ++s) {
            const int row = waveM * 32 + s * 16 + mlo;
            #pragma unroll
            for (int vv = 0; vv < 8; ++vv) {
                const int k = (vv < 4 ? 2 * vv : 8 + 2 * vv) + kh * 8;
                afr[s].u[vv] = *(const unsigned*)&As[row * AS_LD + k];
            }
        }
        #pragma unroll
        for (int s = 0; s < 2; ++s) {
            const int n  = waveN * 32 + s * 16 + mlo;
            const int kb = kh * 8;
            #pragma unroll
            for (int vv = 0; vv < 8; ++vv)
                bfr[s].u[vv] = *(const unsigned*)&Bs[n * BS_LD + kb + (vv < 4 ? 2 * vv : 8 + 2 * vv)];
        }

        // ---- 4 WMMAs: 32x32 per wave per 32-deep K chunk ----
        #pragma unroll
        for (int i = 0; i < 2; ++i)
            #pragma unroll
            for (int j = 0; j < 2; ++j)
                acc[i][j] = __builtin_amdgcn_wmma_f32_16x16x32_f16(
                    false, afr[i].v, false, bfr[j].v,
                    (short)0, acc[i][j], false, false);

        __syncthreads();
    }

    // ---- epilogue: ISA f32 C/D layout (lane lo = N, VGPR r = M row) ----
    // Column-dependent terms hoisted: exactly one bias/mask load per column.
    const int colLo = lane & 15;
    const int rowHi = (lane >> 4) * 8;

    float badd[2] = {0.f, 0.f};
    bool  mz[2]   = {false, false};
    #pragma unroll
    for (int j = 0; j < 2; ++j) {
        const int colB = nBase + waveN * 32 + j * 16 + colLo;
        if (MODE == 0) {
            if (HAS_BIAS) badd[j] = bias[colB];
        } else {
            mz[j] = (Mp[colB] == 0);
        }
    }

    #pragma unroll
    for (int i = 0; i < 2; ++i) {
        #pragma unroll
        for (int j = 0; j < 2; ++j) {
            const int colB = nBase + waveN * 32 + j * 16 + colLo;
            #pragma unroll
            for (int r = 0; r < 8; ++r) {
                const int rowB = mBase + waveM * 32 + i * 16 + rowHi + r;
                float val = acc[i][j][r];
                if (MODE == 0) {
                    val += badd[j];
                } else {
                    val = mz[j] ? -1e-9f : val * scale;
                }
                Op[(size_t)rowB * ldo + colB] = val;
            }
        }
    }
}

// Row softmax over attention [B*H*S rows x 2048]; row kept in registers,
// wave32 shuffle reduction + cross-wave LDS reduction.
__global__ void __launch_bounds__(256)
softmax_rows(float* __restrict__ attn)
{
    const int ROW = S_N;
    const size_t base = (size_t)blockIdx.x * ROW;
    const int tid  = threadIdx.x;
    const int lane = tid & 31;
    const int wave = tid >> 5;

    float r[8];
    #pragma unroll
    for (int i = 0; i < 8; ++i) r[i] = attn[base + tid + i * 256];

    float m = r[0];
    #pragma unroll
    for (int i = 1; i < 8; ++i) m = fmaxf(m, r[i]);
    #pragma unroll
    for (int off = 16; off >= 1; off >>= 1)
        m = fmaxf(m, __shfl_xor(m, off, 32));

    __shared__ float smax[8];
    __shared__ float ssum[8];
    if (lane == 0) smax[wave] = m;
    __syncthreads();
    float bm = smax[0];
    #pragma unroll
    for (int w = 1; w < 8; ++w) bm = fmaxf(bm, smax[w]);

    float s = 0.f;
    #pragma unroll
    for (int i = 0; i < 8; ++i) { r[i] = __expf(r[i] - bm); s += r[i]; }
    #pragma unroll
    for (int off = 16; off >= 1; off >>= 1)
        s += __shfl_xor(s, off, 32);
    if (lane == 0) ssum[wave] = s;
    __syncthreads();
    float bs = 0.f;
    #pragma unroll
    for (int w = 0; w < 8; ++w) bs += ssum[w];

    const float inv = 1.f / bs;
    #pragma unroll
    for (int i = 0; i < 8; ++i) attn[base + tid + i * 256] = r[i] * inv;
}

extern "C" void kernel_launch(void* const* d_in, const int* in_sizes, int n_in,
                              void* d_out, int out_size, void* d_ws, size_t ws_size,
                              hipStream_t stream) {
    (void)in_sizes; (void)n_in; (void)out_size; (void)ws_size;

    const float* q    = (const float*)d_in[0];
    const float* k    = (const float*)d_in[1];
    const float* v    = (const float*)d_in[2];
    const int*   mask = (const int*)  d_in[3];
    const float* Wq   = (const float*)d_in[4];
    const float* bq   = (const float*)d_in[5];
    const float* Wk   = (const float*)d_in[6];
    const float* bk   = (const float*)d_in[7];
    const float* Wv   = (const float*)d_in[8];
    const float* bv   = (const float*)d_in[9];
    const float* Wo   = (const float*)d_in[10];
    const float* bo   = (const float*)d_in[11];

    float* outX = (float*)d_out;                              // [B*S, DM]
    float* attn = outX + (size_t)B_N * S_N * DM_;             // [B,H,S,S]

    float* ws = (float*)d_ws;
    const size_t proj = (size_t)B_N * S_N * DM_;              // 4M floats
    float* Qp = ws;                                           // reused as Xcat later
    float* Kp = ws + proj;
    float* Vp = ws + 2 * proj;
    float* Xc = Qp;

    const long SD  = (long)S_N * DM_;    // per-batch stride of projections
    const long SS  = (long)S_N * S_N;    // per-head stride of attention
    dim3 blk(256);

    // 1) Q/K/V projections: P = X @ W^T + b   (M=4096, N=1024, K=1024)
    dim3 gProj(DM_ / 64, (B_N * S_N) / 128, 1);
    gemm_wmma<0, true, true><<<gProj, blk, 0, stream>>>(q, DM_, 0, 0, Wq, DM_, 0, 0,
        Qp, DM_, 0, 0, bq, nullptr, 0, DM_, 1.f, 1);
    gemm_wmma<0, true, true><<<gProj, blk, 0, stream>>>(k, DM_, 0, 0, Wk, DM_, 0, 0,
        Kp, DM_, 0, 0, bk, nullptr, 0, DM_, 1.f, 1);
    gemm_wmma<0, true, true><<<gProj, blk, 0, stream>>>(v, DM_, 0, 0, Wv, DM_, 0, 0,
        Vp, DM_, 0, 0, bv, nullptr, 0, DM_, 1.f, 1);

    // 2) energy = (Q_h @ K_h^T) / 8, masked -> attention region of d_out
    dim3 gEng(S_N / 64, S_N / 128, B_N * H_N);
    gemm_wmma<1, true, false><<<gEng, blk, 0, stream>>>(
        Qp, DM_, SD, DH_, Kp, DM_, SD, DH_,
        attn, S_N, (long)H_N * SS, SS,
        nullptr, mask, S_N, DH_, 0.125f, H_N);

    // 3) row softmax in place
    softmax_rows<<<dim3(B_N * H_N * S_N), blk, 0, stream>>>(attn);

    // 4) x_h = attn @ V_h, scattered into concat-head layout [B*S, DM]
    dim3 gAV(1, S_N / 128, B_N * H_N);
    gemm_wmma<0, false, false><<<gAV, blk, 0, stream>>>(
        attn, S_N, (long)H_N * SS, SS, Vp, DM_, SD, DH_,
        Xc, DM_, SD, DH_,
        nullptr, nullptr, 0, S_N, 1.f, H_N);

    // 5) out = Xc @ Wo^T + bo -> first region of d_out
    gemm_wmma<0, true, true><<<gProj, blk, 0, stream>>>(Xc, DM_, 0, 0, Wo, DM_, 0, 0,
        outX, DM_, 0, 0, bo, nullptr, 0, DM_, 1.f, 1);
}
